// IntegratedEncoder_46651934769203
// MI455X (gfx1250) — matrix operand order repaired and checked by reading, not compiled, not run on hardware
//
#include <hip/hip_runtime.h>
#include <hip/hip_bf16.h>

#define NN    40000
#define EE    640000
#define FIN   128
#define HIDD  256
#define OUTD  128
#define BG    64

typedef __attribute__((ext_vector_type(16))) __bf16 v16bf;
typedef __attribute__((ext_vector_type(8)))  __bf16 v8bf;
typedef __attribute__((ext_vector_type(8)))  float  v8f;
typedef __attribute__((address_space(3)))    __bf16 lds_bf16_t;

// ---------------- utility kernels ----------------

__global__ void fill0_k(float* __restrict__ p, size_t n) {
  size_t i = (size_t)blockIdx.x * blockDim.x + threadIdx.x;
  if (i < n) p[i] = 0.0f;
}

__global__ void f2bf_k(const float* __restrict__ s, __bf16* __restrict__ d, size_t n) {
  size_t i = (size_t)blockIdx.x * blockDim.x + threadIdx.x;
  if (i < n) d[i] = (__bf16)s[i];
}

__global__ void count_k(const int* __restrict__ idx, float* __restrict__ cnt, int n) {
  int i = blockIdx.x * blockDim.x + threadIdx.x;
  if (i < n) atomicAdd(&cnt[idx[i]], 1.0f);
}

// agg[dst[e]][:] += x[src[e]][:]   (float4-vectorized atomics; node features L2-resident)
__global__ void scatter_add4_k(const float* __restrict__ x, const int* __restrict__ src,
                               const int* __restrict__ dst, float* __restrict__ agg,
                               int E, int d) {
  size_t i = (size_t)blockIdx.x * blockDim.x + threadIdx.x;
  int vec = d >> 2;
  size_t tot = (size_t)E * vec;
  if (i >= tot) return;
  int e = (int)(i / vec);
  int c = (int)(i - (size_t)e * vec) * 4;
  int s = src[e], t = dst[e];
  float4 xv = *(const float4*)(x + (size_t)s * d + c);
  float* o = agg + (size_t)t * d + c;
  atomicAdd(o + 0, xv.x);
  atomicAdd(o + 1, xv.y);
  atomicAdd(o + 2, xv.z);
  atomicAdd(o + 3, xv.w);
}

// out = bf16(agg / max(deg,1))
__global__ void mean_bf_k(const float* __restrict__ agg, const float* __restrict__ deg,
                          __bf16* __restrict__ o, int n, int d) {
  size_t i = (size_t)blockIdx.x * blockDim.x + threadIdx.x;
  size_t tot = (size_t)n * d;
  if (i >= tot) return;
  int node = (int)(i / d);
  o[i] = (__bf16)(agg[i] / fmaxf(deg[node], 1.0f));
}

// gsum[batch[n]][:] += h[n][:]
__global__ void pool_add4_k(const float* __restrict__ h, const int* __restrict__ batch,
                            float* __restrict__ gsum, int n, int d) {
  size_t i = (size_t)blockIdx.x * blockDim.x + threadIdx.x;
  int vec = d >> 2;
  size_t tot = (size_t)n * vec;
  if (i >= tot) return;
  int node = (int)(i / vec);
  int c = (int)(i - (size_t)node * vec) * 4;
  float4 xv = *(const float4*)(h + (size_t)node * d + c);
  float* o = gsum + (size_t)batch[node] * d + c;
  atomicAdd(o + 0, xv.x);
  atomicAdd(o + 1, xv.y);
  atomicAdd(o + 2, xv.z);
  atomicAdd(o + 3, xv.w);
}

__global__ void pool_scale_k(float* __restrict__ gv, const float* __restrict__ gcnt,
                             int b, int d) {
  int i = blockIdx.x * blockDim.x + threadIdx.x;
  if (i >= b * d) return;
  gv[i] = gv[i] / fmaxf(gcnt[i / d], 1.0f);
}

// one wave per row LayerNorm
__global__ void layernorm_rows_k(const float* __restrict__ x, const float* __restrict__ g,
                                 const float* __restrict__ bb, float* __restrict__ y,
                                 int rows, int cols) {
  int wave = threadIdx.x >> 5;
  int lane = threadIdx.x & 31;
  int r = blockIdx.x * (blockDim.x >> 5) + wave;
  if (r >= rows) return;
  const float* xr = x + (size_t)r * cols;
  float s = 0.0f, s2 = 0.0f;
  for (int c = lane; c < cols; c += 32) { float v = xr[c]; s += v; s2 += v * v; }
#pragma unroll
  for (int o = 16; o > 0; o >>= 1) {
    s  += __shfl_xor(s,  o, 32);
    s2 += __shfl_xor(s2, o, 32);
  }
  float mu  = s / cols;
  float var = s2 / cols - mu * mu;
  float inv = rsqrtf(var + 1e-5f);
  for (int c = lane; c < cols; c += 32)
    y[(size_t)r * cols + c] = (xr[c] - mu) * inv * g[c] + bb[c];
}

// out[n][:] = table[batch[n]][:]
__global__ void gather_rows4_k(const float* __restrict__ t, const int* __restrict__ batch,
                               float* __restrict__ out, int n, int d) {
  size_t i = (size_t)blockIdx.x * blockDim.x + threadIdx.x;
  int vec = d >> 2;
  size_t tot = (size_t)n * vec;
  if (i >= tot) return;
  int node = (int)(i / vec);
  int c = (int)(i - (size_t)node * vec);
  ((float4*)out)[i] = ((const float4*)(t + (size_t)batch[node] * d))[c];
}

// ---------------- WMMA fragment helpers ----------------

__device__ __forceinline__ v16bf load_a_frag(const __bf16* __restrict__ row, int kk, int half) {
  // 16-bit A 16x32 layout: lane half selects K-subgroups {0..7,16..23} / {8..15,24..31}
  v8bf lo = *(const v8bf*)(row + kk + half * 8);
  v8bf hi = *(const v8bf*)(row + kk + 16 + half * 8);
  v16bf a;
#pragma unroll
  for (int j = 0; j < 8; ++j) { a[j] = lo[j]; a[8 + j] = hi[j]; }
  return a;
}

__device__ __forceinline__ v16bf load_b_frag(const __bf16* row, int kk, int half) {
  // 16-bit B 32x16: lane half selects K-subgroups {0..7,16..23} / {8..15,24..31} for this column
  v8bf b0 = *(const v8bf*)(row + kk + half * 8);
  v8bf b1 = *(const v8bf*)(row + kk + 16 + half * 8);
  v16bf b;
#pragma unroll
  for (int j = 0; j < 8; ++j) { b[j] = b0[j]; b[8 + j] = b1[j]; }
  return b;
}

// async global -> LDS copy of 16 bytes (CDNA5 ASYNCcnt path)
__device__ __forceinline__ void async_cp16(__bf16* lds_dst, const __bf16* gsrc) {
  unsigned loff = (unsigned)(unsigned long long)(lds_bf16_t*)lds_dst;
  asm volatile("global_load_async_to_lds_b128 %0, %1, off"
               :: "v"(loff), "v"((unsigned long long)gsrc)
               : "memory");
}

// ---------------- fused WMMA GEMM ----------------
// C[M x Nc] = act( A1[M x K] * W1[Nc x K]^T  (+ A2 * W2^T)  + bias )
// Block: 256 thr = 8 waves; block tile BM=128 x BN=32. Each wave owns one
// 16-row M tile and both 16-col N tiles (2 accumulators -> A fragment reused
// 2x, LDS-staged B tiles reused 8x). Weight panels staged once per block via
// async global->LDS DMA; rows padded by 8 bf16 (4-dword skew, conflict-free
// ds_load_b128). Tail-M guard is wave-uniform and sits AFTER the only barrier.
template<int K, bool DUAL>
__global__ __launch_bounds__(256)
void gemm_bf16_wmma_k(const __bf16* __restrict__ A1, const __bf16* __restrict__ W1,
                      const __bf16* __restrict__ A2, const __bf16* __restrict__ W2,
                      const float* __restrict__ bias, float* __restrict__ C,
                      int M, int Nc, int relu) {
  constexpr int NMAT = DUAL ? 2 : 1;
  constexpr int KP = K + 8;
  __shared__ __bf16 Bs[NMAT][32][KP];

  const int tid  = threadIdx.x;
  const int lane = tid & 31;
  const int wv   = tid >> 5;                // 0..7 : M tile within block
  const int r16  = lane & 15;
  const int half = lane >> 4;
  const int nbase = blockIdx.x * 32;

  // ---- stage weight panel(s): NMAT x 32 rows x K, 16B chunks, full EXEC ----
  constexpr int CH = NMAT * 32 * (K / 8);   // multiple of 256 for all configs
  const __bf16* Wm[2] = { W1, DUAL ? W2 : W1 };
#pragma unroll
  for (int c0 = 0; c0 < CH; c0 += 256) {
    int c   = c0 + tid;
    int mat = c / (32 * (K / 8));
    int rem = c - mat * (32 * (K / 8));
    int row = rem / (K / 8);
    int col = (rem - row * (K / 8)) * 8;
    async_cp16(&Bs[mat][row][col], Wm[mat] + (size_t)(nbase + row) * K + col);
  }
  asm volatile("s_wait_asynccnt 0x0" ::: "memory");
  __syncthreads();

  const int tM = blockIdx.y * 8 + wv;
  if (tM * 16 >= M) return;                 // wave-uniform; no barriers after

  const __bf16* a1row = A1 + (size_t)(tM * 16 + r16) * K;
  const __bf16* a2row = DUAL ? (A2 + (size_t)(tM * 16 + r16) * K) : (const __bf16*)0;

  v8f acc0 = {}, acc1 = {};
#pragma unroll
  for (int kk = 0; kk < K; kk += 32) {
    v16bf a1 = load_a_frag(a1row, kk, half);
    v16bf b0 = load_b_frag(&Bs[0][r16][0],      kk, half);
    v16bf b1 = load_b_frag(&Bs[0][16 + r16][0], kk, half);
    acc0 = __builtin_amdgcn_wmma_f32_16x16x32_bf16(false, a1, false, b0, (short)0, acc0,
                                                   false, false);
    acc1 = __builtin_amdgcn_wmma_f32_16x16x32_bf16(false, a1, false, b1, (short)0, acc1,
                                                   false, false);
    if constexpr (DUAL) {
      v16bf a2 = load_a_frag(a2row, kk, half);
      v16bf c0 = load_b_frag(&Bs[1][r16][0],      kk, half);
      v16bf c1 = load_b_frag(&Bs[1][16 + r16][0], kk, half);
      acc0 = __builtin_amdgcn_wmma_f32_16x16x32_bf16(false, a2, false, c0, (short)0, acc0,
                                                     false, false);
      acc1 = __builtin_amdgcn_wmma_f32_16x16x32_bf16(false, a2, false, c1, (short)0, acc1,
                                                     false, false);
    }
    __builtin_prefetch(a1row + kk + 64, 0, 1);   // global_prefetch_b8: next A chunk
  }

  float bs0 = bias[nbase + r16];
  float bs1 = bias[nbase + 16 + r16];
#pragma unroll
  for (int v = 0; v < 8; ++v) {
    int rm = v + 8 * half;                  // C layout: VGPR v -> row v / v+8 by lane half
    size_t ro = (size_t)(tM * 16 + rm) * Nc + nbase;
    float v0 = acc0[v] + bs0;
    float v1 = acc1[v] + bs1;
    if (relu) { v0 = fmaxf(v0, 0.0f); v1 = fmaxf(v1, 0.0f); }
    C[ro + r16]      = v0;
    C[ro + 16 + r16] = v1;
  }
}

// ---------------- launcher ----------------

static inline dim3 gr1(size_t n, int bs) { return dim3((unsigned)((n + bs - 1) / bs)); }

extern "C" void kernel_launch(void* const* d_in, const int* in_sizes, int n_in,
                              void* d_out, int out_size, void* d_ws, size_t ws_size,
                              hipStream_t stream) {
  const float* x_bb   = (const float*)d_in[0];
  const float* x_att  = (const float*)d_in[1];   (void)x_att; // dead: softmax(len1)=1
  const float* W1l    = (const float*)d_in[2];
  const float* b1l    = (const float*)d_in[3];
  const float* W1r    = (const float*)d_in[4];
  const float* W2l    = (const float*)d_in[5];
  const float* b2l    = (const float*)d_in[6];
  const float* W2r    = (const float*)d_in[7];
  const float* W3l    = (const float*)d_in[8];
  const float* b3l    = (const float*)d_in[9];
  const float* W3r    = (const float*)d_in[10];
  const float* Wv     = (const float*)d_in[15];
  const float* bv     = (const float*)d_in[16];
  const float* inpW   = (const float*)d_in[17];
  const float* inpB   = (const float*)d_in[18];
  const float* outW   = (const float*)d_in[19];
  const float* outB   = (const float*)d_in[20];
  const float* ln_g   = (const float*)d_in[21];
  const float* ln_b   = (const float*)d_in[22];
  const int*   eidx   = (const int*)d_in[23];
  const int*   batch  = (const int*)d_in[24];
  const int* src = eidx;
  const int* dst = eidx + EE;

  // ---- workspace carve-out ----
  char* ws = (char*)d_ws;
  size_t off = 0;
  auto carve = [&](size_t bytes) { char* p = ws + off; off = (off + bytes + 255) & ~(size_t)255; return p; };
  float*  deg   = (float*)carve((size_t)NN * 4);
  float*  agg   = (float*)carve((size_t)NN * HIDD * 4);
  float*  hA    = (float*)carve((size_t)NN * HIDD * 4);
  float*  hB    = (float*)carve((size_t)NN * HIDD * 4);
  __bf16* abuf1 = (__bf16*)carve((size_t)NN * HIDD * 2);   // aggregated-mean (bf16)
  __bf16* abuf2 = (__bf16*)carve((size_t)NN * HIDD * 2);   // root features (bf16)
  __bf16* wb    = (__bf16*)carve((size_t)1 << 20);          // converted weights
  float*  gcnt  = (float*)carve((size_t)BG * 4);
  float*  valB  = (float*)carve((size_t)BG * HIDD * 4);
  float*  vvB   = (float*)carve((size_t)BG * HIDD * 4);
  float*  attnB = (float*)carve((size_t)BG * HIDD * 4);
  float*  lnB   = (float*)carve((size_t)BG * HIDD * 4);
  __bf16* gvbf  = (__bf16*)carve((size_t)BG * OUTD * 2);
  __bf16* valbf = (__bf16*)carve((size_t)BG * HIDD * 2);
  __bf16* vvbf  = (__bf16*)carve((size_t)BG * HIDD * 2);
  (void)ws_size;

  __bf16* W1l_b = wb + 0;
  __bf16* W1r_b = wb + 32768;
  __bf16* W2l_b = wb + 65536;
  __bf16* W2r_b = wb + 131072;
  __bf16* W3l_b = wb + 196608;
  __bf16* W3r_b = wb + 229376;
  __bf16* Wv_b  = wb + 262144;
  __bf16* inpV_b= wb + 294912;
  __bf16* outW_b= wb + 360448;

  float* gout = (float*)d_out;              // graph_vec  [B, OUT]
  float* nout = gout + (size_t)BG * OUTD;   // node_embeddings [N, HID]
  (void)out_size; (void)n_in; (void)in_sizes;

  const int BS = 256;
  const unsigned gy = (NN / 16 + 7) / 8;    // 313 blocks of BM=128

  // ---- degrees / graph counts ----
  fill0_k<<<gr1(NN, BS), BS, 0, stream>>>(deg, (size_t)NN);
  fill0_k<<<gr1(BG, BS), BS, 0, stream>>>(gcnt, (size_t)BG);
  count_k<<<gr1(EE, BS), BS, 0, stream>>>(dst, deg, EE);
  count_k<<<gr1(NN, BS), BS, 0, stream>>>(batch, gcnt, NN);

  // ---- weight conversions (f32 -> bf16) ----
  f2bf_k<<<gr1(HIDD * FIN, BS), BS, 0, stream>>>(W1l, W1l_b, (size_t)HIDD * FIN);
  f2bf_k<<<gr1(HIDD * FIN, BS), BS, 0, stream>>>(W1r, W1r_b, (size_t)HIDD * FIN);
  f2bf_k<<<gr1(HIDD * HIDD, BS), BS, 0, stream>>>(W2l, W2l_b, (size_t)HIDD * HIDD);
  f2bf_k<<<gr1(HIDD * HIDD, BS), BS, 0, stream>>>(W2r, W2r_b, (size_t)HIDD * HIDD);
  f2bf_k<<<gr1(OUTD * HIDD, BS), BS, 0, stream>>>(W3l, W3l_b, (size_t)OUTD * HIDD);
  f2bf_k<<<gr1(OUTD * HIDD, BS), BS, 0, stream>>>(W3r, W3r_b, (size_t)OUTD * HIDD);
  f2bf_k<<<gr1(HIDD * OUTD, BS), BS, 0, stream>>>(Wv, Wv_b, (size_t)HIDD * OUTD);
  f2bf_k<<<gr1(HIDD * HIDD, BS), BS, 0, stream>>>(inpW + (size_t)2 * HIDD * HIDD, inpV_b,
                                                  (size_t)HIDD * HIDD);
  f2bf_k<<<gr1(HIDD * HIDD, BS), BS, 0, stream>>>(outW, outW_b, (size_t)HIDD * HIDD);

  // ---- SAGE layer 1: 128 -> 256, relu ----
  fill0_k<<<gr1((size_t)NN * FIN, BS), BS, 0, stream>>>(agg, (size_t)NN * FIN);
  scatter_add4_k<<<gr1((size_t)EE * (FIN / 4), BS), BS, 0, stream>>>(x_bb, src, dst, agg, EE, FIN);
  mean_bf_k<<<gr1((size_t)NN * FIN, BS), BS, 0, stream>>>(agg, deg, abuf1, NN, FIN);
  f2bf_k<<<gr1((size_t)NN * FIN, BS), BS, 0, stream>>>(x_bb, abuf2, (size_t)NN * FIN);
  gemm_bf16_wmma_k<128, true><<<dim3(HIDD / 32, gy), 256, 0, stream>>>(
      abuf1, W1l_b, abuf2, W1r_b, b1l, hA, NN, HIDD, 1);

  // ---- SAGE layer 2: 256 -> 256, relu ----
  fill0_k<<<gr1((size_t)NN * HIDD, BS), BS, 0, stream>>>(agg, (size_t)NN * HIDD);
  scatter_add4_k<<<gr1((size_t)EE * (HIDD / 4), BS), BS, 0, stream>>>(hA, src, dst, agg, EE, HIDD);
  mean_bf_k<<<gr1((size_t)NN * HIDD, BS), BS, 0, stream>>>(agg, deg, abuf1, NN, HIDD);
  f2bf_k<<<gr1((size_t)NN * HIDD, BS), BS, 0, stream>>>(hA, abuf2, (size_t)NN * HIDD);
  gemm_bf16_wmma_k<256, true><<<dim3(HIDD / 32, gy), 256, 0, stream>>>(
      abuf1, W2l_b, abuf2, W2r_b, b2l, hB, NN, HIDD, 1);

  // ---- SAGE layer 3: 256 -> 128, no act ----
  fill0_k<<<gr1((size_t)NN * HIDD, BS), BS, 0, stream>>>(agg, (size_t)NN * HIDD);
  scatter_add4_k<<<gr1((size_t)EE * (HIDD / 4), BS), BS, 0, stream>>>(hB, src, dst, agg, EE, HIDD);
  mean_bf_k<<<gr1((size_t)NN * HIDD, BS), BS, 0, stream>>>(agg, deg, abuf1, NN, HIDD);
  f2bf_k<<<gr1((size_t)NN * HIDD, BS), BS, 0, stream>>>(hB, abuf2, (size_t)NN * HIDD);
  gemm_bf16_wmma_k<256, true><<<dim3(OUTD / 32, gy), 256, 0, stream>>>(
      abuf1, W3l_b, abuf2, W3r_b, b3l, hA, NN, OUTD, 0);

  // ---- global mean pool -> graph_vec (output 0) ----
  fill0_k<<<gr1((size_t)BG * OUTD, BS), BS, 0, stream>>>(gout, (size_t)BG * OUTD);
  pool_add4_k<<<gr1((size_t)NN * (OUTD / 4), BS), BS, 0, stream>>>(hA, batch, gout, NN, OUTD);
  pool_scale_k<<<gr1(BG * OUTD, BS), BS, 0, stream>>>(gout, gcnt, BG, OUTD);

  // ---- attention block (exact: softmax over len-1 == 1, so attn == v per graph) ----
  f2bf_k<<<gr1((size_t)BG * OUTD, BS), BS, 0, stream>>>(gout, gvbf, (size_t)BG * OUTD);
  gemm_bf16_wmma_k<128, false><<<dim3(HIDD / 32, 1), 256, 0, stream>>>(
      gvbf, Wv_b, (const __bf16*)0, (const __bf16*)0, bv, valB, BG, HIDD, 0);
  f2bf_k<<<gr1((size_t)BG * HIDD, BS), BS, 0, stream>>>(valB, valbf, (size_t)BG * HIDD);
  gemm_bf16_wmma_k<256, false><<<dim3(HIDD / 32, 1), 256, 0, stream>>>(
      valbf, inpV_b, (const __bf16*)0, (const __bf16*)0, inpB + 2 * HIDD, vvB, BG, HIDD, 0);
  f2bf_k<<<gr1((size_t)BG * HIDD, BS), BS, 0, stream>>>(vvB, vvbf, (size_t)BG * HIDD);
  gemm_bf16_wmma_k<256, false><<<dim3(HIDD / 32, 1), 256, 0, stream>>>(
      vvbf, outW_b, (const __bf16*)0, (const __bf16*)0, outB, attnB, BG, HIDD, 0);

  // ---- LayerNorm per graph row, then gather per node (output 1) ----
  layernorm_rows_k<<<dim3((BG + 3) / 4), 128, 0, stream>>>(attnB, ln_g, ln_b, lnB, BG, HIDD);
  gather_rows4_k<<<gr1((size_t)NN * (HIDD / 4), BS), BS, 0, stream>>>(lnB, batch, nout, NN, HIDD);
}
